// MultiHeadedAttention_78838419685926
// MI455X (gfx1250) — matrix operand order repaired and checked by reading, not compiled, not run on hardware
//
#include <hip/hip_runtime.h>
#include <stdint.h>

typedef __attribute__((ext_vector_type(16))) __bf16 v16bf;
typedef __attribute__((ext_vector_type(8)))  float  v8f;

#define S_LEN 2048
#define DM    1024
#define NH    16
#define DK    64
#define BATCH 4
#define MTOT  (BATCH * S_LEN)   // 8192 rows

struct U8  { uint32_t x[8]; };   // 32B container for v16bf bit_cast
struct H16 { uint16_t u[16]; };  // 32B container for v16bf bit_cast

// ---- load 16 contiguous bf16 (stored as u16) ----
__device__ __forceinline__ v16bf ld16h(const uint16_t* p) {
  U8 s; const uint32_t* q = (const uint32_t*)p;
#pragma unroll
  for (int i = 0; i < 8; ++i) s.x[i] = q[i];
  return __builtin_bit_cast(v16bf, s);
}
// ---- load two 8-bf16 chunks (A-matrix layout: k=base..+7 and k=base+16..+23) ----
__device__ __forceinline__ v16bf ld8x2h(const uint16_t* lo, const uint16_t* hi) {
  U8 s; const uint32_t* a = (const uint32_t*)lo; const uint32_t* b = (const uint32_t*)hi;
#pragma unroll
  for (int i = 0; i < 4; ++i) { s.x[i] = a[i]; s.x[4 + i] = b[i]; }
  return __builtin_bit_cast(v16bf, s);
}
// ---- fp32 -> bf16 round-to-nearest-even ----
__device__ __forceinline__ uint16_t f2bf(float f) {
  uint32_t u = __builtin_bit_cast(uint32_t, f);
  return (uint16_t)((u + 0x7FFFu + ((u >> 16) & 1u)) >> 16);
}
// ---- pack two f32 C-tiles into one bf16 A-operand (in-register, no LDS) ----
__device__ __forceinline__ v16bf pack2(const v8f& x, const v8f& y) {
  H16 t;
#pragma unroll
  for (int i = 0; i < 8; ++i) { t.u[i] = f2bf(x[i]); t.u[8 + i] = f2bf(y[i]); }
  return __builtin_bit_cast(v16bf, t);
}
__device__ __forceinline__ v8f wmma_bf16(v16bf a, v16bf b, v8f c) {
  // D(f32 16x16) = A(bf16 16x32) x B(bf16 32x16) + C
  return __builtin_amdgcn_wmma_f32_16x16x32_bf16(false, a, false, b, (short)0, c, false, false);
}
__device__ __forceinline__ v8f vzero() {
  v8f v;
#pragma unroll
  for (int i = 0; i < 8; ++i) v[i] = 0.0f;
  return v;
}

// ============================================================================
// Bulk fp32 -> bf16 conversion (done ONCE per tensor; keeps conversion VALU
// out of the WMMA hot loops). 8 elems/thread, b128 in / b128 out.
// ============================================================================
__global__ void __launch_bounds__(256)
cvt_bf16_kernel(const float* __restrict__ in, uint16_t* __restrict__ out, int n) {
  const int idx = (blockIdx.x * 256 + threadIdx.x) * 8;
  if (idx >= n) return;
  float4 a = *(const float4*)(in + idx);
  float4 b = *(const float4*)(in + idx + 4);
  union { uint16_t u[8]; uint4 v; } o;
  o.u[0] = f2bf(a.x); o.u[1] = f2bf(a.y); o.u[2] = f2bf(a.z); o.u[3] = f2bf(a.w);
  o.u[4] = f2bf(b.x); o.u[5] = f2bf(b.y); o.u[6] = f2bf(b.z); o.u[7] = f2bf(b.w);
  *(uint4*)(out + idx) = o.v;
}

// ============================================================================
// Projection GEMM (all-bf16 operands): C[m,n] = sum_k A[m,k]*W[n,k] + bias[n]
// One wave -> 16x64 output tile (4 WMMA accumulators), K-loop step 32.
// MODE 0: Q -> bf16 [B,H,S,DK], post-scaled by (1/sqrt(DK)) * log2(e)
// MODE 1: K -> bf16 [B,H,S,DK]
// MODE 2: V -> bf16 [B,H,DK,S]   (transposed: natural B-operand layout for PV)
// MODE 3: out-proj -> fp32 [B,S,DM]
// ============================================================================
template <int MODE>
__global__ void __launch_bounds__(256)
proj_kernel(const uint16_t* __restrict__ A16, const uint16_t* __restrict__ W16,
            const float* __restrict__ bias, uint16_t* __restrict__ outb,
            float* __restrict__ outf) {
  const int lane = threadIdx.x & 31;
  const int half = lane >> 4;
  const int lq   = lane & 15;
  const int wave = blockIdx.x * 8 + (threadIdx.x >> 5);
  const int m0   = (wave >> 4) * 16;       // 512 m-tiles
  const int n0   = (wave & 15) * 64;       // 16 n-tiles of 64
  const int base = half * 8;               // A-matrix k chunk base

  const uint16_t* pa  = A16 + (size_t)(m0 + lq) * DM;            // A row for lane
  const uint16_t* pw0 = W16 + (size_t)(n0 + lq) * DM + half * 16;

  v8f acc[4];
#pragma unroll
  for (int t = 0; t < 4; ++t) acc[t] = vzero();

  for (int kk = 0; kk < DM; kk += 32) {
    __builtin_prefetch(pa + kk + 256, 0, 1);   // global_prefetch_b8, ~8 iters ahead
    v16bf a = ld16h(pa + kk + half * 16);
#pragma unroll
    for (int t = 0; t < 4; ++t) {
      // B[k][n] = W[n][k]: 16 consecutive k at fixed n -> contiguous bf16
      v16bf b = ld16h(pw0 + (size_t)t * 16 * DM + kk);
      acc[t] = wmma_bf16(a, b, acc[t]);
    }
  }

  const int bidx = m0 >> 11;          // batch (tiles never straddle batches)
  const int sloc = m0 & (S_LEN - 1);
#pragma unroll
  for (int t = 0; t < 4; ++t) {
    const int n = n0 + t * 16 + lq;
    const float bn = bias[n];
#pragma unroll
    for (int i = 0; i < 8; ++i) {
      const int s = sloc + half * 8 + i;  // C layout: row = half*8 + i
      float v = acc[t][i] + bn;
      if (MODE == 0) {
        v *= 0.125f * 1.44269504088896340736f;  // 1/sqrt(64) * log2(e)
        outb[((size_t)(bidx * NH + (n >> 6)) * S_LEN + s) * DK + (n & 63)] = f2bf(v);
      } else if (MODE == 1) {
        outb[((size_t)(bidx * NH + (n >> 6)) * S_LEN + s) * DK + (n & 63)] = f2bf(v);
      } else if (MODE == 2) {
        outb[((size_t)(bidx * NH + (n >> 6)) * DK + (n & 63)) * S_LEN + s] = f2bf(v);
      } else {
        outf[(size_t)(bidx * S_LEN + s) * DM + n] = v;
      }
    }
  }
}

// ============================================================================
// One 64-key flash-attention block, TRANSPOSED scores: St = K x Qt.
//  - St C-layout: lane holds query q = q0+lq (fixed), keys kb+sub*16+half*8+i.
//  - Row max/sum are in-lane + ONE shfl_xor(16) (no 16-lane butterflies).
//  - St C-layout IS the PV A-operand layout: P packs to bf16 in-register,
//    no LDS transpose, no DS waits.
//  - O-accumulator rescale needs per-output-row scale: 8 shfl gathers.
// MASKED only for the single diagonal block (kb = q0 & ~63 covers rows
// q0..q0+15 since q0-kb <= 48); earlier blocks skip all mask ops.
// ============================================================================
template <bool MASKED>
__device__ __forceinline__ void attn_block(
    int kb, int q0, int half, int lq, int base,
    const uint16_t* __restrict__ Kp, const uint16_t* __restrict__ Vp,
    v16bf bq0, v16bf bq1, v8f (&acc)[4], float& mval, float& lval) {
  v8f c[4];
#pragma unroll
  for (int sub = 0; sub < 4; ++sub) {        // 16-key subtiles of St
    const uint16_t* kr = Kp + (size_t)(kb + sub * 16 + lq) * DK;   // A: K rows
    v16bf ka0 = ld16h(kr + half * 16);        // d 0..31
    v16bf ka1 = ld16h(kr + 32 + half * 16);   // d 32..63
    v8f t = vzero();
    t = wmma_bf16(ka0, bq0, t);
    t = wmma_bf16(ka1, bq1, t);
    c[sub] = t;   // c[sub][i] = S[key = kb+sub*16+half*8+i][q = q0+lq]
  }
  if (MASKED) {
    const int q = q0 + lq;
#pragma unroll
    for (int sub = 0; sub < 4; ++sub)
#pragma unroll
      for (int i = 0; i < 8; ++i)
        if (kb + sub * 16 + half * 8 + i > q) c[sub][i] = -1.0e30f;
  }
  // row max: in-lane over 32 keys, then merge other half-wave's 32 keys
  float mx = fmaxf(c[0][0], c[0][1]);
#pragma unroll
  for (int sub = 0; sub < 4; ++sub)
#pragma unroll
    for (int i = (sub == 0 ? 2 : 0); i < 8; ++i) mx = fmaxf(mx, c[sub][i]);
  mx = fmaxf(mx, __shfl_xor(mx, 16, 32));
  const float mn = fmaxf(mval, mx);
  const float sc = exp2f(mval - mn);
  mval = mn;
  // exponentials + row sum (in-lane, then merge other half)
  float rs = 0.0f;
#pragma unroll
  for (int sub = 0; sub < 4; ++sub)
#pragma unroll
    for (int i = 0; i < 8; ++i) {
      float e = exp2f(c[sub][i] - mn);
      c[sub][i] = e;
      rs += e;
    }
  rs += __shfl_xor(rs, 16, 32);
  lval = lval * sc + rs;
  // St C-layout == PV A-operand layout: pack in-register
  const v16bf pa0 = pack2(c[0], c[1]);   // keys kb    .. kb+31
  const v16bf pa1 = pack2(c[2], c[3]);   // keys kb+32 .. kb+63
  // rescale O accumulator: register i holds output row q0+half*8+i, whose
  // scale lives in lane (half*8+i) -> gather
#pragma unroll
  for (int i = 0; i < 8; ++i) {
    const float si = __shfl(sc, half * 8 + i, 32);
#pragma unroll
    for (int t = 0; t < 4; ++t) acc[t][i] *= si;
  }
#pragma unroll
  for (int t = 0; t < 4; ++t) {
    // B[k][d] = Vt[d][k]: contiguous in k at fixed d
    const uint16_t* vr = Vp + (size_t)(t * 16 + lq) * S_LEN + kb;
    acc[t] = wmma_bf16(pa0, ld8x2h(vr + half * 8, vr + half * 8 + 16), acc[t]);
    acc[t] = wmma_bf16(pa1, ld8x2h(vr + 32 + half * 8, vr + 32 + half * 8 + 16), acc[t]);
  }
}

// ============================================================================
// Flash attention: one wave per 16-row q-tile, 64 keys per block, no LDS.
// Steady state runs mask-free; exactly one masked diagonal block at the end.
// ============================================================================
__global__ void __launch_bounds__(256)
attn_kernel(const uint16_t* __restrict__ Qb, const uint16_t* __restrict__ Kb,
            const uint16_t* __restrict__ Vt, uint16_t* __restrict__ Ob) {
  const int lane = threadIdx.x & 31;
  const int wid  = threadIdx.x >> 5;
  const int half = lane >> 4;
  const int lq   = lane & 15;
  const int tile = blockIdx.x * 8 + wid;  // 8192 q-tiles total
  const int qt   = tile & 127;            // q-tile within (b,h)
  const int bh   = tile >> 7;             // 0..63
  const int q0   = qt * 16;

  const uint16_t* Q  = Qb + (size_t)bh * S_LEN * DK;
  const uint16_t* Kp = Kb + (size_t)bh * S_LEN * DK;
  const uint16_t* Vp = Vt + (size_t)bh * DK * S_LEN;

  const int base = half * 8;
  // Qt as B-operand, loaded once: B[d][q], q = lq, d = half*16+e (contiguous)
  const uint16_t* qr = Q + (size_t)(q0 + lq) * DK;
  const v16bf bq0 = ld16h(qr + half * 16);        // d 0..31
  const v16bf bq1 = ld16h(qr + 32 + half * 16);   // d 32..63

  v8f acc[4];
#pragma unroll
  for (int t = 0; t < 4; ++t) acc[t] = vzero();
  float mval = -3.0e38f, lval = 0.0f;

  const int kdiag = q0 & ~63;             // first (and only) block needing masks
  for (int kb = 0; kb < kdiag; kb += 64)
    attn_block<false>(kb, q0, half, lq, base, Kp, Vp, bq0, bq1, acc, mval, lval);
  attn_block<true>(kdiag, q0, half, lq, base, Kp, Vp, bq0, bq1, acc, mval, lval);

  const int bidx = bh >> 4, h = bh & 15;
  const float linv = 1.0f / lval;
#pragma unroll
  for (int i = 0; i < 8; ++i) {
    const float inv = __shfl(linv, half * 8 + i, 32);  // 1/l for row q0+half*8+i
    const int s = q0 + half * 8 + i;
#pragma unroll
    for (int t = 0; t < 4; ++t) {
      const int n = h * DK + t * 16 + lq;
      Ob[(size_t)(bidx * S_LEN + s) * DM + n] = f2bf(acc[t][i] * inv);
    }
  }
}

// ============================================================================
extern "C" void kernel_launch(void* const* d_in, const int* in_sizes, int n_in,
                              void* d_out, int out_size, void* d_ws, size_t ws_size,
                              hipStream_t stream) {
  (void)in_sizes; (void)n_in; (void)out_size; (void)ws_size;
  const float* query = (const float*)d_in[0];
  const float* key   = (const float*)d_in[1];
  const float* value = (const float*)d_in[2];
  const float* Wq = (const float*)d_in[3]; const float* bq = (const float*)d_in[4];
  const float* Wk = (const float*)d_in[5]; const float* bk = (const float*)d_in[6];
  const float* Wv = (const float*)d_in[7]; const float* bv = (const float*)d_in[8];
  const float* Wo = (const float*)d_in[9]; const float* bo = (const float*)d_in[10];
  float* out = (float*)d_out;

  // workspace layout (uint16_t elements):
  //   [0,4HE):   Qb, Kb, Vt, Ab   (bf16 intermediate tensors, 16 MB each)
  //   [4HE,7HE): Xq, Xk, Xv       (bf16-converted activations)
  //   [7HE,..):  Wqb, Wkb, Wvb, Wob (bf16-converted weights, 2 MB each)
  const size_t HE = (size_t)BATCH * NH * S_LEN * DK;  // 8,388,608
  const size_t WE = (size_t)DM * DM;                  // 1,048,576
  uint16_t* base = (uint16_t*)d_ws;
  uint16_t* Qb  = base;
  uint16_t* Kb  = base + HE;
  uint16_t* Vt  = base + 2 * HE;
  uint16_t* Ab  = base + 3 * HE;
  uint16_t* Xq  = base + 4 * HE;
  uint16_t* Xk  = base + 5 * HE;
  uint16_t* Xv  = base + 6 * HE;
  uint16_t* Wqb = base + 7 * HE;
  uint16_t* Wkb = Wqb + WE;
  uint16_t* Wvb = Wkb + WE;
  uint16_t* Wob = Wvb + WE;

  const int actBlocks = (int)(HE / 8 / 256);  // 4096
  const int wBlocks   = (int)(WE / 8 / 256);  // 512
  cvt_bf16_kernel<<<actBlocks, 256, 0, stream>>>(query, Xq, (int)HE);
  cvt_bf16_kernel<<<actBlocks, 256, 0, stream>>>(key,   Xk, (int)HE);
  cvt_bf16_kernel<<<actBlocks, 256, 0, stream>>>(value, Xv, (int)HE);
  cvt_bf16_kernel<<<wBlocks, 256, 0, stream>>>(Wq, Wqb, (int)WE);
  cvt_bf16_kernel<<<wBlocks, 256, 0, stream>>>(Wk, Wkb, (int)WE);
  cvt_bf16_kernel<<<wBlocks, 256, 0, stream>>>(Wv, Wvb, (int)WE);
  cvt_bf16_kernel<<<wBlocks, 256, 0, stream>>>(Wo, Wob, (int)WE);

  const int projBlocks = (MTOT / 16) * (DM / 64) / 8;  // 1024 blocks x 8 waves
  proj_kernel<0><<<projBlocks, 256, 0, stream>>>(Xq, Wqb, bq, Qb, nullptr);
  proj_kernel<1><<<projBlocks, 256, 0, stream>>>(Xk, Wkb, bk, Kb, nullptr);
  proj_kernel<2><<<projBlocks, 256, 0, stream>>>(Xv, Wvb, bv, Vt, nullptr);

  const int attnBlocks = (BATCH * NH * (S_LEN / 16)) / 8;  // 1024 blocks x 8 waves
  attn_kernel<<<attnBlocks, 256, 0, stream>>>(Qb, Kb, Vt, Ab);

  proj_kernel<3><<<projBlocks, 256, 0, stream>>>(Ab, Wob, bo, nullptr, out);
}
